// Router_81114752352734
// MI455X (gfx1250) — matrix operand order repaired and checked by reading, not compiled, not run on hardware
//
#include <hip/hip_runtime.h>
#include <hip/hip_bf16.h>

// ---------------------------------------------------------------------------
// MoE router for MI455X (gfx1250, wave32, WMMA).
//   logits = x@W + noise@Wn   -> top-2 -> renormalized gates
// HBM-bound: 512 MiB activation stream @ 23.3 TB/s ~= 23 us floor.
// Rev 3: double-buffered LDS weight staging (one barrier per chunk, staging
// fully overlapped with WMMA), A-stream software-pipelined one k-tile ahead.
// ---------------------------------------------------------------------------

typedef __attribute__((ext_vector_type(16))) __bf16 v16bf;
typedef __attribute__((ext_vector_type(8)))  float  v8f;

#define D_DIM 4096
#define E_DIM 64
#define M_TOT 16384            // B*S = 4*4096
#define KT 32                  // K per WMMA
#define NTILES 4               // 64/16 N tiles
#define CTOT (D_DIM / KT)      // 128 k-tiles
#define CH 4                   // k-tiles per LDS chunk (K=128 per chunk)
#define NCHUNK (CTOT / CH)     // 32 chunks
#define ROWS_PER_BLOCK 128     // 8 waves * 16 rows
#define THREADS 256
#define LDS_BYTES 32768        // one buffer: 2 mats * 4 ktiles * 4 ntiles * 512 bf16 * 2B
#define WMAT_ELEMS (D_DIM * E_DIM)      // 262144 elements per matrix
#define WMAT_BYTES (WMAT_ELEMS * 2)     // bf16 bytes per packed matrix
#define CHUNK_BYTES_PER_SEL (CH * NTILES * 32 * 16 * 2)  // 16384

// ---------------------------------------------------------------------------
// Pack fp32 weights -> bf16 in WMMA B-fragment order.
// B matrix 32x16 (KxN), wave32 layout (CDNA5 ISA 7.12.2 B pattern):
//   lane l in [0,16):  col N=l,    K = {0..7, 16..23} in register order
//   lane l in [16,32): col N=l-16, K = {8..15, 24..31} in register order
// Packed index: (((c*NTILES + t)*32 + lane)*16 + j).
// ---------------------------------------------------------------------------
__global__ void pack_weights_kernel(const float* __restrict__ w,
                                    const float* __restrict__ nw,
                                    __bf16* __restrict__ out) {
    int o = blockIdx.x * blockDim.x + threadIdx.x;
    if (o >= 2 * WMAT_ELEMS) return;
    int sel  = o / WMAT_ELEMS;
    int r    = o - sel * WMAT_ELEMS;
    int j    = r & 15;
    int lane = (r >> 4) & 31;
    int t    = (r >> 9) & 3;
    int c    = r >> 11;
    int lrow = lane & 15;
    int half = lane >> 4;
    int k    = c * KT + (j < 8 ? half * 8 + j : 16 + half * 8 + (j - 8));
    int n    = t * 16 + lrow;
    const float* src = sel ? nw : w;
    out[o] = (__bf16)src[(size_t)k * E_DIM + n];
}

// Raw fp32 A-fragment for one k-tile: 16 floats per lane.
// A 16x32 layout (ISA 7.12.2): lane half 0 -> K {0..7, 16..23}, half 1 -> +8.
struct RawA { float4 a, b, c, d; };

static __device__ inline RawA load_raw(const float* __restrict__ p) {
    const float4* q = (const float4*)p;
    RawA r;
    r.a = q[0]; r.b = q[1]; r.c = q[4]; r.d = q[5];
    return r;
}

static __device__ inline v16bf cvt16(const RawA& r) {
    v16bf v;
    v[0]  = (__bf16)r.a.x; v[1]  = (__bf16)r.a.y; v[2]  = (__bf16)r.a.z; v[3]  = (__bf16)r.a.w;
    v[4]  = (__bf16)r.b.x; v[5]  = (__bf16)r.b.y; v[6]  = (__bf16)r.b.z; v[7]  = (__bf16)r.b.w;
    v[8]  = (__bf16)r.c.x; v[9]  = (__bf16)r.c.y; v[10] = (__bf16)r.c.z; v[11] = (__bf16)r.c.w;
    v[12] = (__bf16)r.d.x; v[13] = (__bf16)r.d.y; v[14] = (__bf16)r.d.z; v[15] = (__bf16)r.d.w;
    return v;
}

// Half-stage loads: 4 of the 8 per-thread float4s of a chunk.
static __device__ inline void stage_loads(const unsigned char* __restrict__ gbase,
                                          size_t chunk_off, int tid, int it0,
                                          float4 t[4]) {
#pragma unroll
    for (int i = 0; i < 4; ++i) {
        int off    = (it0 + i) * THREADS * 16 + tid * 16;
        int sel    = off >> 14;                       // 16 KB per matrix-chunk
        int within = off & (CHUNK_BYTES_PER_SEL - 1);
        t[i] = *(const float4*)(gbase + (size_t)sel * WMAT_BYTES + chunk_off + within);
    }
}

static __device__ inline void stage_stores(unsigned char* __restrict__ dst,
                                           int tid, int it0, const float4 t[4]) {
#pragma unroll
    for (int i = 0; i < 4; ++i) {
        int off = (it0 + i) * THREADS * 16 + tid * 16;
        *(float4*)(dst + off) = t[i];
    }
}

// One k-tile: 16 ds_load_b128 under one wait, then 8 independent WMMAs.
static __device__ inline void ktile_mma(const unsigned char* __restrict__ cur,
                                        int kk, int lane,
                                        const v16bf& aX, const v16bf& aN,
                                        v8f acc[NTILES]) {
    v16bf bw[NTILES], bn[NTILES];
#pragma unroll
    for (int t = 0; t < NTILES; ++t) {
        bw[t] = *(const v16bf*)(cur + (((0 * CH + kk) * NTILES + t) * 1024) + lane * 32);
        bn[t] = *(const v16bf*)(cur + (((1 * CH + kk) * NTILES + t) * 1024) + lane * 32);
    }
#pragma unroll
    for (int t = 0; t < NTILES; ++t)
        acc[t] = __builtin_amdgcn_wmma_f32_16x16x32_bf16(
                     false, aX, false, bw[t], (short)0, acc[t], false, false);
#pragma unroll
    for (int t = 0; t < NTILES; ++t)
        acc[t] = __builtin_amdgcn_wmma_f32_16x16x32_bf16(
                     false, aN, false, bn[t], (short)0, acc[t], false, false);
}

__global__ __launch_bounds__(THREADS)
void router_kernel(const float* __restrict__ x,
                   const float* __restrict__ noise,
                   const __bf16* __restrict__ packed,
                   float* __restrict__ out_g,
                   int* __restrict__ out_i) {
    __shared__ __align__(32) unsigned char smem[2 * LDS_BYTES];  // double buffer

    const int tid  = threadIdx.x;
    const int wave = tid >> 5;
    const int lane = tid & 31;
    const int half = lane >> 4;
    const int lrow = lane & 15;

    const int rowA = blockIdx.x * ROWS_PER_BLOCK + wave * 16 + lrow;
    const float* xrow = x     + (size_t)rowA * D_DIM + half * 8;
    const float* nrow = noise + (size_t)rowA * D_DIM + half * 8;
    const unsigned char* gbase = (const unsigned char*)packed;

    v8f zero = {};
    v8f acc[NTILES];
#pragma unroll
    for (int t = 0; t < NTILES; ++t) acc[t] = zero;

    // Prime the A-stream pipeline (k-tile 0).
    RawA rx = load_raw(xrow);
    RawA rn = load_raw(nrow);

    // Prologue: stage chunk 0 into buffer 0.
    {
        float4 t[4];
        stage_loads(gbase, 0, tid, 0, t);
        stage_stores(smem, tid, 0, t);
        stage_loads(gbase, 0, tid, 4, t);
        stage_stores(smem, tid, 4, t);
    }
    __syncthreads();

    for (int ch = 0; ch < NCHUNK; ++ch) {
        const unsigned char* cur = smem + (size_t)(ch & 1) * LDS_BYTES;
        unsigned char*       nxt = smem + (size_t)((ch + 1) & 1) * LDS_BYTES;
        const bool have_next = (ch + 1) < NCHUNK;
        const size_t next_off = (size_t)((ch + 1) * CH) * (NTILES * 32 * 16 * 2);

        // First half of next chunk's staging loads, issued before compute.
        float4 ta[4];
        if (have_next) stage_loads(gbase, next_off, tid, 0, ta);

#pragma unroll
        for (int kk = 0; kk < 2; ++kk) {
            RawA rx2{}, rn2{};
            const int cn = ch * CH + kk + 1;
            if (cn < CTOT) { rx2 = load_raw(xrow + cn * KT); rn2 = load_raw(nrow + cn * KT); }
            v16bf aX = cvt16(rx), aN = cvt16(rn);
            ktile_mma(cur, kk, lane, aX, aN, acc);
            rx = rx2; rn = rn2;
        }

        // Retire half A into the idle buffer, launch half B's loads.
        float4 tb[4];
        if (have_next) {
            stage_stores(nxt, tid, 0, ta);
            stage_loads(gbase, next_off, tid, 4, tb);
        }

#pragma unroll
        for (int kk = 2; kk < CH; ++kk) {
            RawA rx2{}, rn2{};
            const int cn = ch * CH + kk + 1;
            if (cn < CTOT) { rx2 = load_raw(xrow + cn * KT); rn2 = load_raw(nrow + cn * KT); }
            v16bf aX = cvt16(rx), aN = cvt16(rn);
            ktile_mma(cur, kk, lane, aX, aN, acc);
            rx = rx2; rn = rn2;
        }

        if (have_next) stage_stores(nxt, tid, 4, tb);
        __syncthreads();   // single barrier per chunk: next buffer is ready
    }

    // Spill logits to LDS buffer 0 (last chunk, 31, was read from buffer 1;
    // the loop's final barrier already separates us from all buffer-0 readers).
    float* logits = (float*)smem;  // [128 rows][64 experts]
#pragma unroll
    for (int t = 0; t < NTILES; ++t)
#pragma unroll
        for (int r = 0; r < 8; ++r)
            logits[(wave * 16 + half * 8 + r) * E_DIM + t * 16 + lrow] = acc[t][r];
    __syncthreads();

    // One thread per row: top-2. Softmax denominator cancels under
    // renormalization: g1 = 1/(1+exp(l2-l1)), g2 = 1-g1.
    if (tid < ROWS_PER_BLOCK) {
        const float* lr = logits + tid * E_DIM;
        float v1 = -INFINITY, v2 = -INFINITY;
        int i1 = 0, i2 = 0;
#pragma unroll 4
        for (int ee = 0; ee < E_DIM; ++ee) {
            int e = (ee + tid) & (E_DIM - 1);   // rotate to dodge bank conflicts
            float v = lr[e];
            bool b1 = (v > v1) || (v == v1 && e < i1);
            bool b2 = (v > v2) || (v == v2 && e < i2);
            if (b1)      { v2 = v1; i2 = i1; v1 = v; i1 = e; }
            else if (b2) { v2 = v;  i2 = e; }
        }
        float tt = __expf(v2 - v1);
        float g1 = 1.0f / (1.0f + tt);
        int gm = blockIdx.x * ROWS_PER_BLOCK + tid;
        out_g[gm * 2 + 0] = g1;
        out_g[gm * 2 + 1] = 1.0f - g1;
        out_i[gm * 2 + 0] = i1;
        out_i[gm * 2 + 1] = i2;
    }
}

extern "C" void kernel_launch(void* const* d_in, const int* in_sizes, int n_in,
                              void* d_out, int out_size, void* d_ws, size_t ws_size,
                              hipStream_t stream) {
    const float* x     = (const float*)d_in[0];
    const float* noise = (const float*)d_in[1];
    const float* w     = (const float*)d_in[2];
    const float* nw    = (const float*)d_in[3];
    // d_in[4] = top_k (always 2 here; kernel specialized for K=2, E=64)

    __bf16* packed = (__bf16*)d_ws;  // 1 MiB: both weight matrices, bf16, B-fragment order

    pack_weights_kernel<<<(2 * WMAT_ELEMS + 255) / 256, 256, 0, stream>>>(w, nw, packed);

    float* out_g = (float*)d_out;
    int*   out_i = (int*)d_out + (size_t)M_TOT * 2;
    router_kernel<<<M_TOT / ROWS_PER_BLOCK, THREADS, 0, stream>>>(
        x, noise, packed, out_g, out_i);
}
